// mLSTMblock_60163901882428
// MI455X (gfx1250) — matrix operand order repaired and checked by reading, not compiled, not run on hardware
//
#include <hip/hip_runtime.h>
#include <math.h>

// ---------------------------------------------------------------------------
// mLSTM block for MI455X (gfx1250, wave32, WMMA 16x16x32 bf16).
// Phase 0: one-time fp32 -> bf16 conversion of x and all weights.
// Phase 1: Wx = x@W^T + b, Ax = x@A^T  (bf16 WMMA, 16x64 strip per wave).
// Phase 2: persistent-grid scan over T (66 WGs = 528 waves = 1 tile/wave),
//          depth-4 rotating fragment pipeline inside each tile GEMM so L2
//          load latency overlaps the WMMA pipe on the serial critical path.
// ---------------------------------------------------------------------------

typedef __attribute__((ext_vector_type(16))) __bf16 v16bf;
typedef __attribute__((ext_vector_type(8)))  __bf16 v8bf;
typedef __attribute__((ext_vector_type(8)))  float  v8f;

#define BSZ   64
#define TLEN  1024
#define DDIM  512
#define HDIM  512
#define RDIM  64
#define FOURH 2048
#define NCOMB (FOURH + RDIM)   // 2112 combined output cols (W then A)

__device__ __forceinline__ int lane_id() { return threadIdx.x & 31; }

// A fragment: 16x32 (MxK) bf16 from row-major bf16 src (leading dim ld).
// ISA 7.12.2: lanes 0-15 -> M=lane, K 0-7 | 16-23; lanes 16-31 -> K 8-15 | 24-31.
__device__ __forceinline__ v16bf load_a_bf(const __bf16* __restrict__ src, int ld) {
  const int lane = lane_id();
  const int row  = lane & 15;
  const int kb   = (lane < 16) ? 0 : 8;
  const __bf16* p = src + (size_t)row * ld + kb;
  v8bf lo = *(const v8bf*)(p);
  v8bf hi = *(const v8bf*)(p + 16);
  return __builtin_shufflevector(lo, hi, 0, 1, 2, 3, 4, 5, 6, 7,
                                         8, 9, 10, 11, 12, 13, 14, 15);
}

// B fragment: 32x16 (KxN) bf16 where logical B[k][n] = w[n*ld + k]
// (row-major weight W[N x K], transposed access; contiguous in K per lane).
// ISA: lanes 0-15 hold K 0-7 | 16-23 (col=lane), lanes 16-31 hold K 8-15 | 24-31.
__device__ __forceinline__ v16bf load_bT_bf(const __bf16* __restrict__ w, int ld) {
  const int lane = lane_id();
  const int col  = lane & 15;
  const int k0   = (lane < 16) ? 0 : 8;
  const __bf16* p = w + (size_t)col * ld + k0;
  v8bf lo = *(const v8bf*)(p);
  v8bf hi = *(const v8bf*)(p + 16);
  return __builtin_shufflevector(lo, hi, 0, 1, 2, 3, 4, 5, 6, 7,
                                         8, 9, 10, 11, 12, 13, 14, 15);
}

__device__ __forceinline__ v8f wmma_bf16(v16bf a, v16bf b, v8f c) {
  return __builtin_amdgcn_wmma_f32_16x16x32_bf16(false, a, false, b, (short)0, c,
                                                 false, false);
}

// Depth-D rotating-buffer 16x16-tile GEMM: C += A(16xK) * W^T(16 rows x K).
// Fully unrolled (K compile-time); slab k+D is loaded right after slab k is
// consumed, forcing D-1 slabs of loads in flight at every WMMA.
template <int K>
__device__ __forceinline__ v8f gemm_tile_deep(const __bf16* __restrict__ a_src,
                                              const __bf16* __restrict__ w_src,
                                              int ld, v8f acc) {
  constexpr int NS = K / 32;
  constexpr int D  = (NS < 4) ? NS : 4;
  v16bf aB[D], bB[D];
#pragma unroll
  for (int i = 0; i < D; ++i) {
    aB[i] = load_a_bf(a_src + i * 32, ld);
    bB[i] = load_bT_bf(w_src + i * 32, ld);
  }
#pragma unroll
  for (int k = 0; k < NS; ++k) {
    acc = wmma_bf16(aB[k % D], bB[k % D], acc);
    if (k + D < NS) {
      aB[k % D] = load_a_bf(a_src + (k + D) * 32, ld);
      bB[k % D] = load_bT_bf(w_src + (k + D) * 32, ld);
    }
  }
  return acc;
}

// ---------------------------------------------------------------------------
// Phase 0: fp32 -> bf16 (grid-stride)
// ---------------------------------------------------------------------------
__global__ void cvt_kernel(const float* __restrict__ src,
                           __bf16* __restrict__ dst, int n) {
  int i = (int)(blockIdx.x * blockDim.x + threadIdx.x);
  const int stride = (int)(gridDim.x * blockDim.x);
  for (; i < n; i += stride) dst[i] = (__bf16)src[i];
}

// ---------------------------------------------------------------------------
// Phase 1: one 16x64 output strip per wave (A-fragment reused 4x), depth-2
// rotating slab pipeline (occupancy provides the rest of the latency hiding).
// Combined N space: 33 quads (quads 0..31 -> Wx with bias, quad 32 -> Ax).
// ---------------------------------------------------------------------------
__global__ __launch_bounds__(256) void proj_kernel(
    const __bf16* __restrict__ xbf, const __bf16* __restrict__ Wcbf,
    const float* __restrict__ W_b, float* __restrict__ Wx,
    float* __restrict__ Ax) {
  const int wid   = (int)((blockIdx.x * blockDim.x + threadIdx.x) >> 5);
  const int NQ    = NCOMB / 64;                 // 33
  const int total = (BSZ * TLEN / 16) * NQ;     // 4096 * 33
  if (wid >= total) return;
  const int mt = wid / NQ;
  const int q  = wid % NQ;

  const __bf16* arow = xbf  + (size_t)mt * 16 * DDIM;
  const __bf16* wrow = Wcbf + (size_t)q * 64 * DDIM;
  v8f acc[4] = {};

  v16bf aB[2], bB[2][4];
#pragma unroll
  for (int i = 0; i < 2; ++i) {
    aB[i] = load_a_bf(arow + i * 32, DDIM);
#pragma unroll
    for (int s = 0; s < 4; ++s)
      bB[i][s] = load_bT_bf(wrow + (size_t)s * 16 * DDIM + i * 32, DDIM);
  }

  constexpr int NS = DDIM / 32;                 // 16 slabs
#pragma unroll
  for (int k = 0; k < NS; ++k) {
    if ((k & 3) == 0) {                         // gfx1250 global_prefetch_b8
      __builtin_prefetch(arow + k * 32 + 128, 0, 1);
      __builtin_prefetch(wrow + k * 32 + 128, 0, 1);
    }
#pragma unroll
    for (int s = 0; s < 4; ++s) acc[s] = wmma_bf16(aB[k & 1], bB[k & 1][s], acc[s]);
    if (k + 2 < NS) {
      aB[k & 1] = load_a_bf(arow + (k + 2) * 32, DDIM);
#pragma unroll
      for (int s = 0; s < 4; ++s)
        bB[k & 1][s] = load_bT_bf(wrow + (size_t)s * 16 * DDIM + (k + 2) * 32, DDIM);
    }
  }

  const int lane = lane_id();
  const int col  = lane & 15;
  const int m0   = (lane < 16) ? 0 : 8;
#pragma unroll
  for (int s = 0; s < 4; ++s) {
    const int ncol = q * 64 + s * 16 + col;     // 0..2111
    if (ncol < FOURH) {
      const float bias = W_b[ncol];
#pragma unroll
      for (int v = 0; v < 8; ++v) {
        const size_t row = (size_t)mt * 16 + m0 + v;
        Wx[row * FOURH + ncol] = acc[s][v] + bias;
      }
    } else {
#pragma unroll
      for (int v = 0; v < 8; ++v) {
        const size_t row = (size_t)mt * 16 + m0 + v;
        Ax[row * RDIM + (ncol - FOURH)] = acc[s][v];
      }
    }
  }
}

// ---------------------------------------------------------------------------
// Device-wide sense-reversal barrier (persistent grid).
// ---------------------------------------------------------------------------
__device__ void grid_barrier(unsigned* bar, int nblocks) {
  __syncthreads();
  if (threadIdx.x == 0) {
    unsigned* cnt = bar;
    unsigned* gen = bar + 1;
    __threadfence();
    const unsigned g = __hip_atomic_load(gen, __ATOMIC_ACQUIRE, __HIP_MEMORY_SCOPE_AGENT);
    const unsigned a = __hip_atomic_fetch_add(cnt, 1u, __ATOMIC_ACQ_REL, __HIP_MEMORY_SCOPE_AGENT);
    if (a == (unsigned)nblocks - 1u) {
      __hip_atomic_store(cnt, 0u, __ATOMIC_RELAXED, __HIP_MEMORY_SCOPE_AGENT);
      __hip_atomic_fetch_add(gen, 1u, __ATOMIC_ACQ_REL, __HIP_MEMORY_SCOPE_AGENT);
    } else {
      while (__hip_atomic_load(gen, __ATOMIC_ACQUIRE, __HIP_MEMORY_SCOPE_AGENT) == g)
        __builtin_amdgcn_s_sleep(2);
    }
    __threadfence();
  }
  __syncthreads();
}

__device__ __forceinline__ float sigmoidf_(float v) {
  return 1.0f / (1.0f + __expf(-v));
}

// ---------------------------------------------------------------------------
// Phase 2: persistent scan kernel. gridDim.x MUST equal SCAN_WGS.
// 66 WGs * 8 waves = 528 waves: exactly one stage-1 tile per wave per step.
// Few waves per SIMD -> allow max registers (no occupancy to protect).
// ---------------------------------------------------------------------------
#define SCAN_WGS 66

__global__ __launch_bounds__(256)
__attribute__((amdgpu_waves_per_eu(1)))
void scan_kernel(
    const float* __restrict__ Wx,   const float* __restrict__ Ax,
    const __bf16* __restrict__ Ubf, const float* __restrict__ U_b,
    const __bf16* __restrict__ Bbf, const __bf16* __restrict__ Pbf,
    float* __restrict__ gates, float* __restrict__ rbuf,
    __bf16* __restrict__ ubf,  float* __restrict__ mixbuf,
    float* __restrict__ hbuf,  __bf16* __restrict__ hbf,
    float* __restrict__ cbuf,  unsigned* bar, float* __restrict__ out) {
  const int NB       = (int)gridDim.x;              // SCAN_WGS
  const int nthreads = NB * (int)blockDim.x;        // 16896
  const int nwaves   = nthreads >> 5;               // 528
  const int gtid     = (int)(blockIdx.x * blockDim.x + threadIdx.x);
  const int gwid     = gtid >> 5;
  const int lane     = threadIdx.x & 31;
  const int col      = lane & 15;
  const int m0       = (lane < 16) ? 0 : 8;

  for (int t = 0; t < TLEN; ++t) {
    // ---- stage 1: gates_pre = h@U^T (+Ub+Wx_t) [512 tiles]; r = h@B^T [16]
    for (int job = gwid; job < 528; job += nwaves) {
      if (job < 512) {
        const int gm = job >> 7, gn = job & 127;
        v8f acc = {};
        acc = gemm_tile_deep<HDIM>(hbf + (size_t)gm * 16 * HDIM,
                                   Ubf + (size_t)gn * 16 * HDIM, HDIM, acc);
        const int n = gn * 16 + col;
        const float ub = U_b[n];
#pragma unroll
        for (int v = 0; v < 8; ++v) {
          const int b = gm * 16 + m0 + v;
          gates[b * FOURH + n] =
              acc[v] + ub + Wx[((size_t)b * TLEN + t) * FOURH + n];
        }
      } else {
        const int rj = job - 512;
        const int rm = rj >> 2, rn = rj & 3;
        v8f acc = {};
        acc = gemm_tile_deep<HDIM>(hbf + (size_t)rm * 16 * HDIM,
                                   Bbf + (size_t)rn * 16 * HDIM, HDIM, acc);
        const int j = rn * 16 + col;
#pragma unroll
        for (int v = 0; v < 8; ++v) rbuf[(rm * 16 + m0 + v) * RDIM + j] = acc[v];
      }
    }
    grid_barrier(bar, NB);

    // ---- stage 2: u = Ax_t * r  (64x64), stored as bf16 for the mix GEMM
    for (int i = gtid; i < BSZ * RDIM; i += nthreads) {
      const int b = i >> 6, j = i & 63;
      ubf[i] = (__bf16)(Ax[((size_t)b * TLEN + t) * RDIM + j] * rbuf[i]);
    }
    grid_barrier(bar, NB);

    // ---- stage 3: mix = u @ P^T  (64x512, K=64) [128 tiles]
    for (int job = gwid; job < 128; job += nwaves) {
      const int gm = job >> 5, gn = job & 31;
      v8f acc = {};
      acc = gemm_tile_deep<RDIM>(ubf + (size_t)gm * 16 * RDIM,
                                 Pbf + (size_t)gn * 16 * RDIM, RDIM, acc);
#pragma unroll
      for (int v = 0; v < 8; ++v)
        mixbuf[(gm * 16 + m0 + v) * HDIM + gn * 16 + col] = acc[v];
    }
    grid_barrier(bar, NB);

    // ---- stage 4: cell update + emit h (fp32 out, bf16 shadow for GEMMs)
    for (int i = gtid; i < BSZ * HDIM; i += nthreads) {
      const int b = i >> 9, m = i & 511;
      const float* g = gates + (size_t)b * FOURH;
      const float si = sigmoidf_(g[m]);
      const float sf = sigmoidf_(g[HDIM + m]);
      const float so = sigmoidf_(g[2 * HDIM + m]);
      const float tg = tanhf(g[3 * HDIM + m]);
      const float c  = sf * cbuf[i] + si * tg + 0.1f * mixbuf[i];
      const float h  = so * tanhf(c);
      cbuf[i] = c;
      hbuf[i] = h;
      hbf[i]  = (__bf16)h;
      out[((size_t)b * TLEN + t) * HDIM + m] = h;
    }
    grid_barrier(bar, NB);   // protect h before next step's stage 1
  }

  // final (h, c) appended after outs
  const size_t tail = (size_t)BSZ * TLEN * HDIM;
  for (int i = gtid; i < BSZ * HDIM; i += nthreads) {
    out[tail + i]              = hbuf[i];
    out[tail + BSZ * HDIM + i] = cbuf[i];
  }
}

// ---------------------------------------------------------------------------
// Per-call deterministic init of h, c and barrier state.
// ---------------------------------------------------------------------------
__global__ void init_kernel(float* __restrict__ hbuf, __bf16* __restrict__ hbf,
                            float* __restrict__ cbuf, unsigned* __restrict__ bar) {
  const int i = (int)(blockIdx.x * blockDim.x + threadIdx.x);
  if (i < BSZ * HDIM) { hbuf[i] = 0.0f; cbuf[i] = 0.0f; hbf[i] = (__bf16)0.0f; }
  if (i < 2) bar[i] = 0u;
}

// ---------------------------------------------------------------------------
extern "C" void kernel_launch(void* const* d_in, const int* in_sizes, int n_in,
                              void* d_out, int out_size, void* d_ws, size_t ws_size,
                              hipStream_t stream) {
  const float* x   = (const float*)d_in[0];
  const float* W_w = (const float*)d_in[1];
  const float* W_b = (const float*)d_in[2];
  const float* U_w = (const float*)d_in[3];
  const float* U_b = (const float*)d_in[4];
  const float* A_w = (const float*)d_in[5];
  const float* B_w = (const float*)d_in[6];
  const float* P_w = (const float*)d_in[7];
  float* out = (float*)d_out;

  // Workspace layout (units: float = 4 bytes; all blocks 16B aligned)
  float* ws     = (float*)d_ws;
  float* gates  = ws;                                     // 64*2048
  float* rbuf   = gates  + (size_t)BSZ * FOURH;           // 64*64
  float* mixbuf = rbuf   + (size_t)BSZ * RDIM;            // 64*512
  float* hbuf   = mixbuf + (size_t)BSZ * HDIM;            // 64*512
  float* cbuf   = hbuf   + (size_t)BSZ * HDIM;            // 64*512
  unsigned* bar = (unsigned*)(cbuf + (size_t)BSZ * HDIM); // 2 u32, pad 64 f
  float* Ax     = cbuf + (size_t)BSZ * HDIM + 64;         // 65536*64
  float* Wx     = Ax   + (size_t)BSZ * TLEN * RDIM;       // 65536*2048
  __bf16* xbf   = (__bf16*)(Wx + (size_t)BSZ * TLEN * FOURH); // 65536*512
  __bf16* Wcbf  = xbf  + (size_t)BSZ * TLEN * DDIM;       // 2112*512 (W_w|A_w)
  __bf16* Ubf   = Wcbf + (size_t)NCOMB * DDIM;            // 2048*512
  __bf16* Bbf   = Ubf  + (size_t)FOURH * HDIM;            // 64*512
  __bf16* Pbf   = Bbf  + (size_t)RDIM * HDIM;             // 512*64
  __bf16* hbf   = Pbf  + (size_t)HDIM * RDIM;             // 64*512
  __bf16* ubf   = hbf  + (size_t)BSZ * HDIM;              // 64*64

  // 0) init h/c/barrier
  init_kernel<<<(BSZ * HDIM + 255) / 256, 256, 0, stream>>>(hbuf, hbf, cbuf, bar);

  // 0b) one-time fp32 -> bf16 of all GEMM operands
  cvt_kernel<<<4096, 256, 0, stream>>>(x,   xbf,  BSZ * TLEN * DDIM);
  cvt_kernel<<<1024, 256, 0, stream>>>(W_w, Wcbf, FOURH * DDIM);
  cvt_kernel<<<128,  256, 0, stream>>>(A_w, Wcbf + (size_t)FOURH * DDIM, RDIM * DDIM);
  cvt_kernel<<<1024, 256, 0, stream>>>(U_w, Ubf,  FOURH * HDIM);
  cvt_kernel<<<128,  256, 0, stream>>>(B_w, Bbf,  RDIM * HDIM);
  cvt_kernel<<<128,  256, 0, stream>>>(P_w, Pbf,  HDIM * RDIM);

  // 1) parallel projections: 16x64 strip per wave, 8 waves/block
  {
    const int quads  = (BSZ * TLEN / 16) * (NCOMB / 64);  // 4096*33
    const int blocks = (quads + 7) / 8;
    proj_kernel<<<blocks, 256, 0, stream>>>(xbf, Wcbf, W_b, Wx, Ax);
  }

  // 2) persistent sequential scan
  scan_kernel<<<SCAN_WGS, 256, 0, stream>>>(Wx, Ax, Ubf, U_b, Bbf, Pbf,
                                            gates, rbuf, ubf, mixbuf,
                                            hbuf, hbf, cbuf, bar, out);
}
